// SAGPool_40604620816784
// MI455X (gfx1250) — matrix operand order repaired and checked by reading, not compiled, not run on hardware
//
#include <hip/hip_runtime.h>
#include <hip/hip_bf16.h>
#include <math.h>

// ---------------------------------------------------------------------------
// SAGPool pipeline for MI455X (gfx1250, wave32).
//   conv1:   h0 = x @ W1          -> WMMA f16 (v_wmma_f32_16x16x32_f16)
//            deg/agg via edge atomics (L2-resident: h tables ~6.4MB << 192MB L2)
//            h = relu(agg + h0/deg + b1)
//   score:   s = h@pw1 + (segment_sum h[src] -> dst)@pw2 + pb
//   top-K:   exact, deterministic: ordered-uint key, 64Ki-bin histogram,
//            descending prefix bases, bucket scatter, per-bucket sort
//            by (key desc, index asc)  == jax.lax.top_k tie-break/order.
//   conv2:   h2 = hp @ w2 (tiny), masked edge agg on new indices,
//            out = log_softmax(relu(agg2 + h2/deg2 + b2))
//
// Round-1 fix: W1 kept TRANSPOSED in LDS ([h][k]) so each lane's WMMA
// B-fragment is 16 contiguous halves -> two ds_load_b128 instead of
// sixteen ds_load_u16 + merges per fragment.
// ---------------------------------------------------------------------------

typedef __attribute__((ext_vector_type(16))) _Float16 v16h;
typedef __attribute__((ext_vector_type(8)))  _Float16 v8h;
typedef __attribute__((ext_vector_type(8)))  float    v8f;

#define F_IN 128
#define HID  16
#define NCLS 10
#define NBUCK 65536

// ------------------------------ utility ------------------------------------
__global__ void zero_i32(int* p, long n) {
  long i = (long)blockIdx.x * blockDim.x + threadIdx.x;
  for (; i < n; i += (long)gridDim.x * blockDim.x) p[i] = 0;
}
__global__ void zero_f32(float* p, long n) {
  long i = (long)blockIdx.x * blockDim.x + threadIdx.x;
  for (; i < n; i += (long)gridDim.x * blockDim.x) p[i] = 0.0f;
}

// --------------------------- conv1 GEMM (WMMA) ------------------------------
// One wave computes a 16(row)x16(HID) tile of h0 = x @ W1, K=128 in 4 chunks.
__global__ __launch_bounds__(256)
void k_gemm1(const float* __restrict__ x, const float* __restrict__ w1,
             float* __restrict__ h0, int N) {
  // W1 transposed in LDS: w1sT[h*F_IN + k] = W1[k][h]  (4 KB)
  __shared__ _Float16 w1sT[HID * F_IN];
  for (int i = threadIdx.x; i < F_IN * HID; i += 256) {
    int hcol = i >> 7, k = i & 127;
    w1sT[i] = (_Float16)w1[k * HID + hcol];
  }
  __syncthreads();

  const int wave = threadIdx.x >> 5;
  const int lane = threadIdx.x & 31;
  const int hh   = lane >> 4;                    // lane half (0: lanes 0-15)
  const int r    = lane & 15;
  const int tiles = (N + 15) >> 4;
  const int tile  = blockIdx.x * 8 + wave;
  if (tile >= tiles) return;

  const int row0 = tile << 4;
  int row = row0 + r; if (row >= N) row = N - 1;     // clamp (loads only)

  // B fragments: b[i] = W1[kc*32 + 16*hh + i][r]  -> contiguous in w1sT:
  //   w1sT[r*F_IN + kc*32 + 16*hh + i], 32B-aligned -> 2x ds_load_b128 each
  v16h bf[4];
  #pragma unroll
  for (int kc = 0; kc < 4; ++kc) {
    const v8h* bp = (const v8h*)(w1sT + r * F_IN + kc * 32 + 16 * hh);
    v8h lo = bp[0], hi = bp[1];
    bf[kc] = __builtin_shufflevector(lo, hi, 0, 1, 2, 3, 4, 5, 6, 7,
                                     8, 9, 10, 11, 12, 13, 14, 15);
  }

  v8f c = {};
  const float* xr = x + (long)row * F_IN;
  #pragma unroll
  for (int kc = 0; kc < 4; ++kc) {
    // A fragment: rows = lane&15; K = kc*32 + 16*hh + i (matches B order)
    v16h a;
    const float* p0 = xr + kc * 32 + 16 * hh;
    #pragma unroll
    for (int i = 0; i < 8; ++i)  a[i]     = (_Float16)p0[i];
    #pragma unroll
    for (int i = 0; i < 8; ++i)  a[8 + i] = (_Float16)p0[8 + i];
    c = __builtin_amdgcn_wmma_f32_16x16x32_f16(false, a, false, bf[kc],
                                               (short)0, c, false, false);
  }
  // D layout: lane half hh, vreg v -> D[row0 + v + 8*hh][col = r]
  #pragma unroll
  for (int v = 0; v < 8; ++v) {
    int rw = row0 + v + 8 * hh;
    if (rw < N) h0[(long)rw * HID + r] = c[v];
  }
}

// ------------------------------ degree / norm ------------------------------
__global__ void k_count_dst(const int* __restrict__ dst, int* __restrict__ cnt,
                            long E) {
  long e = (long)blockIdx.x * blockDim.x + threadIdx.x;
  if (e < E) atomicAdd(&cnt[dst[e]], 1);
}
__global__ void k_dinv(const int* __restrict__ cnt, float* __restrict__ dinv,
                       int n) {
  int i = blockIdx.x * blockDim.x + threadIdx.x;
  if (i < n) dinv[i] = rsqrtf((float)cnt[i] + 1.0f);
}

// ------------------------- conv1 edge aggregation --------------------------
__global__ void k_agg1(const int* __restrict__ src, const int* __restrict__ dst,
                       const float* __restrict__ dinv,
                       const float* __restrict__ h0, float* __restrict__ agg,
                       long E) {
  long t = (long)blockIdx.x * blockDim.x + threadIdx.x;
  if (t >= E * HID) return;
  long e = t >> 4; int f = (int)(t & 15);
  int s = src[e], d = dst[e];
  float w = dinv[s] * dinv[d];
  atomicAdd(&agg[(long)d * HID + f], w * h0[(long)s * HID + f]);
}
__global__ void k_finish1(const float* __restrict__ agg,
                          const float* __restrict__ h0,
                          const int* __restrict__ cnt,
                          const float* __restrict__ b1,
                          float* __restrict__ h, int N) {
  long t = (long)blockIdx.x * blockDim.x + threadIdx.x;
  if (t >= (long)N * HID) return;
  long n = t >> 4; int f = (int)(t & 15);
  float deg = (float)cnt[n] + 1.0f;
  float v = agg[t] + h0[t] / deg + b1[f];
  h[t] = v > 0.0f ? v : 0.0f;
}

// ------------------------------ score GNN ----------------------------------
__global__ void k_aggH(const int* __restrict__ src, const int* __restrict__ dst,
                       const float* __restrict__ h, float* __restrict__ aggH,
                       long E) {
  long t = (long)blockIdx.x * blockDim.x + threadIdx.x;
  if (t >= E * HID) return;
  long e = t >> 4; int f = (int)(t & 15);
  atomicAdd(&aggH[(long)dst[e] * HID + f], h[(long)src[e] * HID + f]);
}
__global__ void k_score(const float* __restrict__ h,
                        const float* __restrict__ aggH,
                        const float* __restrict__ pw1,
                        const float* __restrict__ pw2,
                        const float* __restrict__ pb,
                        float* __restrict__ s, unsigned* __restrict__ key,
                        int N) {
  int n = blockIdx.x * blockDim.x + threadIdx.x;
  if (n >= N) return;
  float acc = pb[0];
  #pragma unroll
  for (int f = 0; f < HID; ++f)
    acc += h[(long)n * HID + f] * pw1[f] + aggH[(long)n * HID + f] * pw2[f];
  s[n] = acc;
  unsigned u = __float_as_uint(acc);                 // order-preserving key
  key[n] = (u & 0x80000000u) ? ~u : (u | 0x80000000u);
}

// ---------------------------- exact top-K ----------------------------------
__global__ void k_hist(const unsigned* __restrict__ key, int* __restrict__ hist,
                       int N) {
  int n = blockIdx.x * blockDim.x + threadIdx.x;
  if (n < N) atomicAdd(&hist[key[n] >> 16], 1);
}
// descending bases: startD[b] = #keys in buckets strictly greater than b
__global__ __launch_bounds__(256)
void k_scan_desc(const int* __restrict__ hist, int* __restrict__ startD) {
  __shared__ int chunk[256];
  int t = threadIdx.x;
  int sum = 0;
  for (int j = 0; j < 256; ++j) sum += hist[t * 256 + j];
  chunk[t] = sum;
  __syncthreads();
  if (t == 0) {
    int acc = 0;
    for (int i = 255; i >= 0; --i) { int c = chunk[i]; chunk[i] = acc; acc += c; }
  }
  __syncthreads();
  int acc = chunk[t];
  for (int b = t * 256 + 255; b >= t * 256; --b) { startD[b] = acc; acc += hist[b]; }
}
__global__ void k_scatter(const unsigned* __restrict__ key,
                          const int* __restrict__ startD, int* __restrict__ cnt,
                          int* __restrict__ perm, int N) {
  int n = blockIdx.x * blockDim.x + threadIdx.x;
  if (n >= N) return;
  int b = key[n] >> 16;
  int pos = startD[b] + atomicAdd(&cnt[b], 1);
  perm[pos] = n;
}
// per-bucket selection sort by (key desc, index asc) -> deterministic perm
__global__ void k_sort_buckets(const unsigned* __restrict__ key,
                               const int* __restrict__ hist,
                               const int* __restrict__ startD,
                               int* __restrict__ perm) {
  int b = blockIdx.x * blockDim.x + threadIdx.x;
  if (b >= NBUCK) return;
  int m = hist[b];
  if (m < 2) return;
  int st = startD[b];
  for (int i = 0; i < m - 1; ++i) {
    int best = i;
    int bn = perm[st + best]; unsigned bk = key[bn];
    for (int j = i + 1; j < m; ++j) {
      int nj = perm[st + j]; unsigned kj = key[nj];
      if (kj > bk || (kj == bk && nj < bn)) { best = j; bk = kj; bn = nj; }
    }
    int tmp = perm[st + i]; perm[st + i] = perm[st + best]; perm[st + best] = tmp;
  }
}

// ------------------------- pooling / re-index ------------------------------
__global__ void k_pool(const int* __restrict__ perm, const float* __restrict__ h,
                       const float* __restrict__ s, int* __restrict__ kept,
                       int* __restrict__ nidx, float* __restrict__ hp, int K) {
  int i = blockIdx.x * blockDim.x + threadIdx.x;
  if (i >= K) return;
  int node = perm[i];
  kept[node] = 1;
  nidx[node] = i;
  float t = tanhf(s[node]);
  #pragma unroll
  for (int f = 0; f < HID; ++f)
    hp[(long)i * HID + f] = h[(long)node * HID + f] * t;
}

// ------------------------------- conv2 -------------------------------------
__global__ __launch_bounds__(256)
void k_gemm2(const float* __restrict__ hp, const float* __restrict__ w2,
             float* __restrict__ h2, int K) {
  __shared__ float w2s[HID * NCLS];
  for (int i = threadIdx.x; i < HID * NCLS; i += 256) w2s[i] = w2[i];
  __syncthreads();
  long t = (long)blockIdx.x * blockDim.x + threadIdx.x;
  if (t >= (long)K * NCLS) return;
  long i = t / NCLS; int c = (int)(t % NCLS);
  float acc = 0.0f;
  #pragma unroll
  for (int f = 0; f < HID; ++f) acc += hp[i * HID + f] * w2s[f * NCLS + c];
  h2[t] = acc;
}
__global__ void k_deg2(const int* __restrict__ src, const int* __restrict__ dst,
                       const int* __restrict__ kept, const int* __restrict__ nidx,
                       int* __restrict__ cnt2, long E) {
  long e = (long)blockIdx.x * blockDim.x + threadIdx.x;
  if (e >= E) return;
  int s = src[e], d = dst[e];
  if (kept[s] && kept[d]) atomicAdd(&cnt2[nidx[d]], 1);
}
__global__ void k_agg2(const int* __restrict__ src, const int* __restrict__ dst,
                       const int* __restrict__ kept, const int* __restrict__ nidx,
                       const float* __restrict__ dinv2,
                       const float* __restrict__ h2, float* __restrict__ agg2,
                       long E) {
  long t = (long)blockIdx.x * blockDim.x + threadIdx.x;
  if (t >= E * NCLS) return;
  long e = t / NCLS; int c = (int)(t % NCLS);
  int s = src[e], d = dst[e];
  if (!(kept[s] && kept[d])) return;
  int ns = nidx[s], nd = nidx[d];
  float w = dinv2[ns] * dinv2[nd];
  atomicAdd(&agg2[(long)nd * NCLS + c], w * h2[(long)ns * NCLS + c]);
}
__global__ void k_out(const float* __restrict__ agg2,
                      const float* __restrict__ h2,
                      const int* __restrict__ cnt2,
                      const float* __restrict__ b2,
                      float* __restrict__ out, int K) {
  int i = blockIdx.x * blockDim.x + threadIdx.x;
  if (i >= K) return;
  float deg = (float)cnt2[i] + 1.0f;
  float v[NCLS];
  float mx = -1e30f;
  #pragma unroll
  for (int c = 0; c < NCLS; ++c) {
    float u = agg2[(long)i * NCLS + c] + h2[(long)i * NCLS + c] / deg + b2[c];
    u = u > 0.0f ? u : 0.0f;
    v[c] = u;
    mx = u > mx ? u : mx;
  }
  float se = 0.0f;
  #pragma unroll
  for (int c = 0; c < NCLS; ++c) se += __expf(v[c] - mx);
  float lse = logf(se);
  #pragma unroll
  for (int c = 0; c < NCLS; ++c) out[(long)i * NCLS + c] = v[c] - mx - lse;
}

// ------------------------------- launcher ----------------------------------
static inline size_t alignup(size_t x) { return (x + 255) & ~(size_t)255; }

extern "C" void kernel_launch(void* const* d_in, const int* in_sizes, int n_in,
                              void* d_out, int out_size, void* d_ws, size_t ws_size,
                              hipStream_t stream) {
  const float* x   = (const float*)d_in[0];
  const int*   ei  = (const int*)d_in[1];
  const float* w1  = (const float*)d_in[2];
  const float* b1  = (const float*)d_in[3];
  const float* pw1 = (const float*)d_in[4];
  const float* pw2 = (const float*)d_in[5];
  const float* pb  = (const float*)d_in[6];
  const float* w2  = (const float*)d_in[7];
  const float* b2  = (const float*)d_in[8];
  float* out = (float*)d_out;

  const int  N = in_sizes[0] / F_IN;
  const long E = (long)in_sizes[1] / 2;
  const int  K = (N + 1) / 2;
  const int* src = ei;
  const int* dst = ei + E;

  // ---- workspace carve-out ----
  char* p = (char*)d_ws; size_t off = 0;
  auto take = [&](size_t bytes) { void* q = p + off; off += alignup(bytes); return q; };
  float*    h0    = (float*)   take((size_t)N * HID * 4);
  float*    h     = (float*)   take((size_t)N * HID * 4);
  float*    agg   = (float*)   take((size_t)N * HID * 4);
  float*    aggH  = (float*)   take((size_t)N * HID * 4);
  int*      cnt1  = (int*)     take((size_t)N * 4);
  float*    dinv  = (float*)   take((size_t)N * 4);
  float*    s     = (float*)   take((size_t)N * 4);
  unsigned* key   = (unsigned*)take((size_t)N * 4);
  int*      hist  = (int*)     take((size_t)NBUCK * 4);
  int*      stD   = (int*)     take((size_t)NBUCK * 4);
  int*      bcnt  = (int*)     take((size_t)NBUCK * 4);
  int*      perm  = (int*)     take((size_t)N * 4);
  int*      kept  = (int*)     take((size_t)N * 4);
  int*      nidx  = (int*)     take((size_t)N * 4);
  float*    hp    = (float*)   take((size_t)K * HID * 4);
  float*    h2    = (float*)   take((size_t)K * NCLS * 4);
  int*      cnt2  = (int*)     take((size_t)K * 4);
  float*    dinv2 = (float*)   take((size_t)K * 4);
  float*    agg2  = (float*)   take((size_t)K * NCLS * 4);
  (void)ws_size; (void)n_in; (void)out_size;

  const int T = 256;
  auto blk = [&](long n) { return (int)((n + T - 1) / T); };

  // ---- zero-init everything that accumulates (fresh each call) ----
  zero_i32<<<blk(N), T, 0, stream>>>(cnt1, N);
  zero_f32<<<blk((long)N * HID), T, 0, stream>>>(agg, (long)N * HID);
  zero_f32<<<blk((long)N * HID), T, 0, stream>>>(aggH, (long)N * HID);
  zero_i32<<<blk(NBUCK), T, 0, stream>>>(hist, NBUCK);
  zero_i32<<<blk(NBUCK), T, 0, stream>>>(bcnt, NBUCK);
  zero_i32<<<blk(N), T, 0, stream>>>(kept, N);
  zero_i32<<<blk(N), T, 0, stream>>>(nidx, N);
  zero_i32<<<blk(K), T, 0, stream>>>(cnt2, K);
  zero_f32<<<blk((long)K * NCLS), T, 0, stream>>>(agg2, (long)K * NCLS);

  // ---- conv1 ----
  const int tiles = (N + 15) / 16;
  k_gemm1<<<(tiles + 7) / 8, 256, 0, stream>>>(x, w1, h0, N);
  k_count_dst<<<blk(E), T, 0, stream>>>(dst, cnt1, E);
  k_dinv<<<blk(N), T, 0, stream>>>(cnt1, dinv, N);
  k_agg1<<<blk(E * HID), T, 0, stream>>>(src, dst, dinv, h0, agg, E);
  k_finish1<<<blk((long)N * HID), T, 0, stream>>>(agg, h0, cnt1, b1, h, N);

  // ---- scores ----
  k_aggH<<<blk(E * HID), T, 0, stream>>>(src, dst, h, aggH, E);
  k_score<<<blk(N), T, 0, stream>>>(h, aggH, pw1, pw2, pb, s, key, N);

  // ---- exact top-K ----
  k_hist<<<blk(N), T, 0, stream>>>(key, hist, N);
  k_scan_desc<<<1, 256, 0, stream>>>(hist, stD);
  k_scatter<<<blk(N), T, 0, stream>>>(key, stD, bcnt, perm, N);
  k_sort_buckets<<<blk(NBUCK), T, 0, stream>>>(key, hist, stD, perm);
  k_pool<<<blk(K), T, 0, stream>>>(perm, h, s, kept, nidx, hp, K);

  // ---- conv2 + log_softmax ----
  k_gemm2<<<blk((long)K * NCLS), T, 0, stream>>>(hp, w2, h2, K);
  k_deg2<<<blk(E), T, 0, stream>>>(src, dst, kept, nidx, cnt2, E);
  k_dinv<<<blk(K), T, 0, stream>>>(cnt2, dinv2, K);
  k_agg2<<<blk(E * NCLS), T, 0, stream>>>(src, dst, kept, nidx, dinv2, h2, agg2, E);
  k_out<<<blk(K), T, 0, stream>>>(agg2, h2, cnt2, b2, out, K);
}
